// SABlock_37529424232889
// MI455X (gfx1250) — compile-verified, matrix operand order checked by reading, not run on hardware
//
#include <hip/hip_runtime.h>

// ---------------------------------------------------------------------------
// Problem constants (fixed by the reference): B=4, T=2048, C=1024, H=16, D=64
// ---------------------------------------------------------------------------
namespace {
constexpr int Bc = 4, Tc = 2048, Cc = 1024, Hc = 16, Dc = 64;
constexpr int Mrows = Bc * Tc;   // 8192
constexpr int Kdim  = Cc;        // 1024
constexpr int Ncols = Cc;        // 1024
}

#define DEVINL __device__ __forceinline__

typedef __attribute__((ext_vector_type(16))) __bf16 bf16x16;
typedef __attribute__((ext_vector_type(8)))  float  f32x8;
typedef __attribute__((ext_vector_type(4)))  int    i32x4;

union Frag {
  uint4   u[2];   // two 16-byte chunks = 16 bf16
  bf16x16 v;
};

// --------- optional gfx1250 async global->LDS path (guarded) ---------------
#if defined(__has_builtin)
#if __has_builtin(__builtin_amdgcn_global_load_async_to_lds_b128)
#define USE_ASYNC_LDS 1
#endif
#endif

#if defined(USE_ASYNC_LDS)
// Builtin parameter types (from the round-2 diagnostic): pointers to int4
// vectors; global side is address_space(1) ("__device__"), LDS side is
// address_space(3). AS3 pointers are 32-bit: the low 32 bits of the flat
// LDS-aperture address are the LDS byte offset (ISA aperture table).
typedef __attribute__((address_space(1))) i32x4 g_i32x4;
typedef __attribute__((address_space(3))) i32x4 l_i32x4;
DEVINL g_i32x4* to_g(const void* p) { return (g_i32x4*)(uintptr_t)p; }
DEVINL l_i32x4* to_l(void* p) { return (l_i32x4*)(unsigned)(uintptr_t)p; }
DEVINL void wait_async0() {
#if __has_builtin(__builtin_amdgcn_s_wait_asynccnt)
  __builtin_amdgcn_s_wait_asynccnt(0);
#else
  asm volatile("s_wait_asynccnt 0x0" ::: "memory");
#endif
}
#endif

DEVINL __bf16 to_bf16(float f) {
  union { float f; unsigned u; } x; x.f = f;
  unsigned r = x.u + 0x7FFFu + ((x.u >> 16) & 1u);   // round-to-nearest-even
  union { unsigned short s; __bf16 b; } y; y.s = (unsigned short)(r >> 16);
  return y.b;
}

DEVINL f32x8 wmma_bf16(bf16x16 a, bf16x16 b, f32x8 c) {
  // D = A(16x32 bf16) x B(32x16 bf16) + C(16x16 f32)
  return __builtin_amdgcn_wmma_f32_16x16x32_bf16(
      /*neg_a=*/false, a, /*neg_b=*/false, b,
      /*c_mod=*/(short)0, c, /*reuse_a=*/false, /*reuse_b=*/false);
}

// ---------------------------------------------------------------------------
// f32 -> bf16 cast (4 elements / thread)
// ---------------------------------------------------------------------------
__global__ void __launch_bounds__(256)
cast_kernel(const float* __restrict__ in, __bf16* __restrict__ out, int n) {
  const int i = (blockIdx.x * 256 + threadIdx.x) * 4;
  if (i + 3 < n) {
    const float4 f = *(const float4*)(in + i);
    union { ushort4 s; __bf16 b[4]; } u;
    u.b[0] = to_bf16(f.x); u.b[1] = to_bf16(f.y);
    u.b[2] = to_bf16(f.z); u.b[3] = to_bf16(f.w);
    *(ushort4*)(out + i) = u.s;
  }
}

// ---------------------------------------------------------------------------
// GEMM: Y[M,N] = A[M,K] @ W[N,K]^T  (both operands K-contiguous bf16)
// Each wave register-blocks a 32x32 tile: 2 A-frags x 2 B-frags -> 4 WMMAs
// per 32-wide K step (2x fragment reuse vs 16x16).
// Block: 256 threads = 8 waves in a 2(M) x 4(N) arrangement -> 64x128 tile.
// Grid: (M/64, N/128)
// mode 0/1/2: write bf16 scattered into [B,H,T,D]; mode 0 scales by 1/sqrt(D)
// mode 3:     write f32 row-major [M,N] with bias (final output projection)
// ---------------------------------------------------------------------------
__global__ void __launch_bounds__(256)
gemm_bf16_kernel(const __bf16* __restrict__ A, const __bf16* __restrict__ W,
                 const float* __restrict__ bias, __bf16* __restrict__ out_bf,
                 float* __restrict__ out_f32, int mode) {
  const int tid  = threadIdx.x;
  const int lane = tid & 31;
  const int wid  = tid >> 5;
  const int nl   = lane & 15;   // M/N index within a 16-wide tile
  const int hi   = lane >> 4;   // which K-half this lane holds

  const int mbase = blockIdx.x * 64  + (wid >> 2) * 32;
  const int nbase = blockIdx.y * 128 + (wid & 3) * 32;

  const __bf16* arow0 = A + (size_t)(mbase + nl) * Kdim;
  const __bf16* arow1 = arow0 + (size_t)16 * Kdim;
  const __bf16* wrow0 = W + (size_t)(nbase + nl) * Kdim;
  const __bf16* wrow1 = wrow0 + (size_t)16 * Kdim;

  f32x8 acc[2][2];
#pragma unroll
  for (int mi = 0; mi < 2; ++mi)
#pragma unroll
    for (int ni = 0; ni < 2; ++ni) acc[mi][ni] = (f32x8){0, 0, 0, 0, 0, 0, 0, 0};

#pragma unroll 2
  for (int kb = 0; kb < Kdim; kb += 32) {
    // speculative prefetch of the stream ahead (gfx1250 global_prefetch_b8)
    __builtin_prefetch((const void*)(arow0 + kb + 512), 0, 0);
    __builtin_prefetch((const void*)(wrow0 + kb + 512), 0, 0);

    Frag af0, af1, wf0, wf1;
    af0.u[0] = *(const uint4*)(arow0 + kb + hi * 8);
    af0.u[1] = *(const uint4*)(arow0 + kb + 16 + hi * 8);
    af1.u[0] = *(const uint4*)(arow1 + kb + hi * 8);
    af1.u[1] = *(const uint4*)(arow1 + kb + 16 + hi * 8);
    wf0.u[0] = *(const uint4*)(wrow0 + kb + hi * 8);
    wf0.u[1] = *(const uint4*)(wrow0 + kb + 16 + hi * 8);
    wf1.u[0] = *(const uint4*)(wrow1 + kb + hi * 8);
    wf1.u[1] = *(const uint4*)(wrow1 + kb + 16 + hi * 8);

    acc[0][0] = wmma_bf16(af0.v, wf0.v, acc[0][0]);
    acc[0][1] = wmma_bf16(af0.v, wf1.v, acc[0][1]);
    acc[1][0] = wmma_bf16(af1.v, wf0.v, acc[1][0]);
    acc[1][1] = wmma_bf16(af1.v, wf1.v, acc[1][1]);
  }

  // C layout: VGPR r, lanes 0-15 -> M = r, lanes 16-31 -> M = 8 + r; N = lane&15
#pragma unroll
  for (int mi = 0; mi < 2; ++mi) {
#pragma unroll
    for (int ni = 0; ni < 2; ++ni) {
#pragma unroll
      for (int r = 0; r < 8; ++r) {
        const int mm = mbase + mi * 16 + r + 8 * hi;
        const int nn = nbase + ni * 16 + nl;
        float v = acc[mi][ni][r];
        if (mode == 3) {
          out_f32[(size_t)mm * Ncols + nn] = v + bias[nn];
        } else {
          if (mode == 0) v *= 0.125f;  // 1/sqrt(D), prescale Q
          const int b = mm >> 11, t = mm & (Tc - 1);
          const int h = nn >> 6,  d = nn & (Dc - 1);
          out_bf[(((size_t)(b * Hc + h)) * Tc + t) * Dc + d] = to_bf16(v);
        }
      }
    }
  }
}

// ---------------------------------------------------------------------------
// Flash attention (causal, online softmax).
// Grid: (T/128, B*H). Block 256 = 8 waves; wave w owns queries [q0w, q0w+16).
// Key blocks of 32 staged in LDS (async global->LDS when available),
// V kept transposed in LDS so P@V B-fragments are contiguous ds_load_b128.
// ---------------------------------------------------------------------------
__global__ void __launch_bounds__(256)
attn_kernel(const __bf16* __restrict__ Q, const __bf16* __restrict__ K,
            const __bf16* __restrict__ V, __bf16* __restrict__ Y) {
  __shared__ __align__(16) __bf16 Kls[32 * 64];      // [k_local][d]
  __shared__ __align__(16) __bf16 Vls[64 * 32];      // transposed: [d][k_local]
  __shared__ __align__(16) __bf16 Pls[8 * 16 * 32];  // per-wave P staging
#if defined(USE_ASYNC_LDS)
  __shared__ __align__(16) __bf16 Vst[32 * 64];      // row-major V stage
#endif

  const int tid  = threadIdx.x;
  const int lane = tid & 31;
  const int wid  = tid >> 5;
  const int nl   = lane & 15;
  const int hi   = lane >> 4;

  const int bh     = blockIdx.y;          // b*H + h
  const int q_base = blockIdx.x * 128;
  const int q0w    = q_base + wid * 16;

  const __bf16* Qp = Q + (size_t)bh * Tc * Dc;
  const __bf16* Kp = K + (size_t)bh * Tc * Dc;
  const __bf16* Vp = V + (size_t)bh * Tc * Dc;

  // Q A-fragments: 16x64 split into two 16x32 K-chunks, kept in registers.
  Frag qf[2];
#pragma unroll
  for (int c = 0; c < 2; ++c) {
    const __bf16* qrow = Qp + (size_t)(q0w + nl) * Dc + c * 32 + hi * 8;
    qf[c].u[0] = *(const uint4*)(qrow);
    qf[c].u[1] = *(const uint4*)(qrow + 16);
  }

  f32x8 O[4];
#pragma unroll
  for (int dt = 0; dt < 4; ++dt) O[dt] = (f32x8){0, 0, 0, 0, 0, 0, 0, 0};
  float row_max[8], row_sum[8];
#pragma unroll
  for (int r = 0; r < 8; ++r) { row_max[r] = -1e30f; row_sum[r] = 0.0f; }

  const int kb_end = q_base + 128;               // causal bound (uniform per block)
  for (int kb = 0; kb < kb_end; kb += 32) {
    __syncthreads();                             // previous tile fully consumed
    const int kr = tid >> 3;                     // 0..31 : key row in block
    const int c0 = (tid & 7) * 8;                // 0..56 : d offset
#if defined(USE_ASYNC_LDS)
    // Async DMA into LDS: no VGPR roundtrip, tracked by ASYNCcnt.
    __builtin_amdgcn_global_load_async_to_lds_b128(
        to_g(Kp + (size_t)(kb + kr) * Dc + c0), to_l(&Kls[kr * 64 + c0]), 0, 0);
    __builtin_amdgcn_global_load_async_to_lds_b128(
        to_g(Vp + (size_t)(kb + kr) * Dc + c0), to_l(&Vst[kr * 64 + c0]), 0, 0);
    wait_async0();
    __syncthreads();
    {
      // LDS->LDS transpose of the V tile
      const uint4 vv = *(const uint4*)&Vst[kr * 64 + c0];
      const __bf16* ve = (const __bf16*)&vv;
#pragma unroll
      for (int i = 0; i < 8; ++i) Vls[(c0 + i) * 32 + kr] = ve[i];
    }
    __syncthreads();
#else
    {
      const uint4 kk = *(const uint4*)(Kp + (size_t)(kb + kr) * Dc + c0);
      *(uint4*)&Kls[kr * 64 + c0] = kk;
      const uint4 vv = *(const uint4*)(Vp + (size_t)(kb + kr) * Dc + c0);
      const __bf16* ve = (const __bf16*)&vv;
#pragma unroll
      for (int i = 0; i < 8; ++i) Vls[(c0 + i) * 32 + kr] = ve[i];
    }
    __syncthreads();
#endif

    if (kb > q0w + 15) continue;                 // keys entirely in this wave's future

    // Scores S[16 x 32] as two 16x16 tiles, K-dim (=D=64) in two WMMA steps.
    f32x8 s[2];
#pragma unroll
    for (int j = 0; j < 2; ++j) {
      Frag kf0, kf1;
      const __bf16* krow = &Kls[(j * 16 + nl) * 64 + hi * 8];
      kf0.u[0] = *(const uint4*)(krow);
      kf0.u[1] = *(const uint4*)(krow + 16);
      kf1.u[0] = *(const uint4*)(krow + 32);
      kf1.u[1] = *(const uint4*)(krow + 48);
      f32x8 z = (f32x8){0, 0, 0, 0, 0, 0, 0, 0};
      z    = wmma_bf16(qf[0].v, kf0.v, z);
      s[j] = wmma_bf16(qf[1].v, kf1.v, z);
    }

    // Causal mask + online softmax update; P staged to LDS as bf16.
#pragma unroll
    for (int r = 0; r < 8; ++r) {
      const int qi = q0w + r + 8 * hi;
      float v0 = (kb + nl)      <= qi ? s[0][r] : -1e30f;
      float v1 = (kb + 16 + nl) <= qi ? s[1][r] : -1e30f;
      float mx = fmaxf(v0, v1);                  // row max over 16 lanes (half-wave)
      mx = fmaxf(mx, __shfl_xor(mx, 1));
      mx = fmaxf(mx, __shfl_xor(mx, 2));
      mx = fmaxf(mx, __shfl_xor(mx, 4));
      mx = fmaxf(mx, __shfl_xor(mx, 8));
      const float mnew  = fmaxf(row_max[r], mx);
      const float alpha = __expf(row_max[r] - mnew);
      row_max[r] = mnew;
      const float p0 = __expf(v0 - mnew);
      const float p1 = __expf(v1 - mnew);
      row_sum[r] = row_sum[r] * alpha + p0 + p1; // per-lane partial; reduced at end
#pragma unroll
      for (int dt = 0; dt < 4; ++dt) O[dt][r] *= alpha;
      __bf16* prow = &Pls[(wid * 16 + r + 8 * hi) * 32];
      prow[nl]      = to_bf16(p0);
      prow[16 + nl] = to_bf16(p1);
    }

    // O += P(16x32) @ V(32x64): P re-read in A layout, V^T gives contiguous frags.
    Frag pf;
    {
      const __bf16* prow = &Pls[(wid * 16 + nl) * 32 + hi * 8];
      pf.u[0] = *(const uint4*)(prow);
      pf.u[1] = *(const uint4*)(prow + 16);
    }
#pragma unroll
    for (int dt = 0; dt < 4; ++dt) {
      Frag vf;
      const __bf16* vrow = &Vls[(dt * 16 + nl) * 32 + hi * 8];
      vf.u[0] = *(const uint4*)(vrow);
      vf.u[1] = *(const uint4*)(vrow + 16);
      O[dt] = wmma_bf16(pf.v, vf.v, O[dt]);
    }
  }

  // Normalize and store as bf16 rows of [B*T, C] for the output projection.
  const int b = bh >> 4, h = bh & 15;
#pragma unroll
  for (int r = 0; r < 8; ++r) {
    float l = row_sum[r];
    l += __shfl_xor(l, 1);
    l += __shfl_xor(l, 2);
    l += __shfl_xor(l, 4);
    l += __shfl_xor(l, 8);
    const float inv = 1.0f / l;
    const int t = q0w + r + 8 * hi;
    __bf16* yrow = Y + ((size_t)(b * Tc + t)) * Cc + h * Dc;
#pragma unroll
    for (int dt = 0; dt < 4; ++dt)
      yrow[dt * 16 + nl] = to_bf16(O[dt][r] * inv);
  }
}

// ---------------------------------------------------------------------------
// Launcher
// ---------------------------------------------------------------------------
extern "C" void kernel_launch(void* const* d_in, const int* in_sizes, int n_in,
                              void* d_out, int out_size, void* d_ws, size_t ws_size,
                              hipStream_t stream) {
  const float* x  = (const float*)d_in[0];
  const float* Wq = (const float*)d_in[1];
  const float* Wk = (const float*)d_in[2];
  const float* Wv = (const float*)d_in[3];
  const float* Wo = (const float*)d_in[4];
  const float* bo = (const float*)d_in[5];
  float* out = (float*)d_out;

  char* ws = (char*)d_ws;
  constexpr size_t XBF = (size_t)Mrows * Kdim * 2;  // 16 MiB per activation tensor
  constexpr size_t WBF = (size_t)Ncols * Kdim * 2;  // 2 MiB per weight tensor
  __bf16* x_bf  = (__bf16*)(ws);
  __bf16* Wq_bf = (__bf16*)(ws + XBF);
  __bf16* Wk_bf = (__bf16*)(ws + XBF + 1 * WBF);
  __bf16* Wv_bf = (__bf16*)(ws + XBF + 2 * WBF);
  __bf16* Wo_bf = (__bf16*)(ws + XBF + 3 * WBF);
  __bf16* q_ws  = (__bf16*)(ws + 1 * XBF + 4 * WBF);  // [B,H,T,D], pre-scaled
  __bf16* k_ws  = (__bf16*)(ws + 2 * XBF + 4 * WBF);  // [B,H,T,D]
  __bf16* v_ws  = (__bf16*)(ws + 3 * XBF + 4 * WBF);  // [B,H,T,D]
  __bf16* y_bf  = (__bf16*)(ws + 4 * XBF + 4 * WBF);  // attention out, [B*T, C]

  const int nx = Mrows * Kdim;  // 8,388,608
  const int nw = Ncols * Kdim;  // 1,048,576
  cast_kernel<<<nx / 1024, 256, 0, stream>>>(x,  x_bf,  nx);
  cast_kernel<<<nw / 1024, 256, 0, stream>>>(Wq, Wq_bf, nw);
  cast_kernel<<<nw / 1024, 256, 0, stream>>>(Wk, Wk_bf, nw);
  cast_kernel<<<nw / 1024, 256, 0, stream>>>(Wv, Wv_bf, nw);
  cast_kernel<<<nw / 1024, 256, 0, stream>>>(Wo, Wo_bf, nw);

  const dim3 ggrid(Mrows / 64, Ncols / 128);
  gemm_bf16_kernel<<<ggrid, 256, 0, stream>>>(x_bf, Wq_bf, nullptr, q_ws, nullptr, 0);
  gemm_bf16_kernel<<<ggrid, 256, 0, stream>>>(x_bf, Wk_bf, nullptr, k_ws, nullptr, 1);
  gemm_bf16_kernel<<<ggrid, 256, 0, stream>>>(x_bf, Wv_bf, nullptr, v_ws, nullptr, 2);

  attn_kernel<<<dim3(Tc / 128, Bc * Hc), 256, 0, stream>>>(q_ws, k_ws, v_ws, y_bf);

  gemm_bf16_kernel<<<ggrid, 256, 0, stream>>>(y_bf, Wo_bf, bo, nullptr, out, 3);
}
